// LlamaBlockNSA_75874892251627
// MI455X (gfx1250) — matrix operand order repaired and checked by reading, not compiled, not run on hardware
//
#include <hip/hip_runtime.h>
#include <hip/hip_bf16.h>

typedef __bf16 bf16;
typedef __attribute__((ext_vector_type(16))) __bf16 v16bf;
typedef __attribute__((ext_vector_type(8)))  float  v8f;

// ---- static config (matches reference) ----
#define S_LEN 2048
#define DIM   1024
#define NHEAD 16
#define NGRP  4
#define HPG   4
#define DK    64
#define NC    127     // compressed blocks
#define NB    32      // selection blocks (S/64)
#define NSEL  16
#define WWIN  512
#define LDQ   (NHEAD*DK)   // 1024
#define LDKV  (NGRP*DK)    // 256
#define GATE_LD 128        // padded gate row stride (true cols = 48)

// =====================================================================
// elementwise converts
// =====================================================================
__global__ void f32_to_bf16_kernel(const float* __restrict__ src,
                                   bf16* __restrict__ dst, int n) {
  int i = blockIdx.x * 256 + threadIdx.x;
  if (i < n) dst[i] = (bf16)src[i];
}

// Wg [1024,48] f32 -> padded [1024,128] bf16 (cols 48..127 zero)
__global__ void convpad_wg_kernel(const float* __restrict__ src,
                                  bf16* __restrict__ dst) {
  int i = blockIdx.x * 256 + threadIdx.x;
  if (i >= DIM * GATE_LD) return;
  int kk = i >> 7, n = i & (GATE_LD - 1);
  dst[i] = (n < 3 * NHEAD) ? (bf16)src[kk * (3 * NHEAD) + n] : (bf16)0.0f;
}

__global__ void add_out_kernel(const float* __restrict__ a,
                               const float* __restrict__ b,
                               float* __restrict__ o, int n) {
  int i = blockIdx.x * 256 + threadIdx.x;
  if (i < n) o[i] = a[i] + b[i];
}

// =====================================================================
// RMSNorm (optionally fused residual-add): y_bf16 = rms(x+dlt)*w, x1=x+dlt
// =====================================================================
__global__ __launch_bounds__(256) void rmsnorm_kernel(
    const float* __restrict__ X, const float* __restrict__ Dlt,
    const float* __restrict__ w, float* __restrict__ X1out,
    bf16* __restrict__ Yb, int cols) {
  __shared__ float red[256];
  const int row = blockIdx.x;
  const float* xr = X + (size_t)row * cols;
  const float* dr = Dlt ? (Dlt + (size_t)row * cols) : nullptr;
  float ss = 0.f;
  for (int c = threadIdx.x; c < cols; c += 256) {
    float v = xr[c] + (dr ? dr[c] : 0.f);
    ss += v * v;
  }
  red[threadIdx.x] = ss;
  __syncthreads();
  for (int off = 128; off > 0; off >>= 1) {
    if (threadIdx.x < off) red[threadIdx.x] += red[threadIdx.x + off];
    __syncthreads();
  }
  float inv = rsqrtf(red[0] / (float)cols + 1e-6f);
  for (int c = threadIdx.x; c < cols; c += 256) {
    float v = xr[c] + (dr ? dr[c] : 0.f);
    if (X1out) X1out[(size_t)row * cols + c] = v;
    Yb[(size_t)row * cols + c] = (bf16)(v * inv * w[c]);
  }
}

// =====================================================================
// bf16 WMMA GEMM: C[M,N] = A[M,K](bf16,row) @ B[K,N](bf16,row)
// tile 128x128x32, 256 threads = 8 waves, wave owns 16x128 strip.
// REQUIRES M%128==0, N%128==0, K%32==0 (guaranteed by all call sites).
// mode: 0 = f32 out, 1 = bf16 out, 2 = silu -> bf16 out
// All fragment LDS loads issued before the 8-WMMA burst so the loads
// pipeline behind one dscnt wait and WMMAs issue back-to-back.
// =====================================================================
__global__ __launch_bounds__(256) void gemm_bf16_wmma(
    const bf16* __restrict__ A, const bf16* __restrict__ B,
    void* __restrict__ Cout, int M, int N, int K, int mode) {
  __shared__ bf16 As[128][40];   // [m][k]
  __shared__ bf16 Bt[128][40];   // [n][k] (transposed stage)
  const int tid  = threadIdx.x;
  const int wave = tid >> 5;
  const int lane = tid & 31;
  const int lrow = lane & 15;
  const int hi   = lane >> 4;          // 0: lanes 0-15, 1: lanes 16-31
  const int m_blk = blockIdx.y * 128;
  const int n_blk = blockIdx.x * 128;
  const int m0 = wave * 16;

  // staging assignments (fixed per thread)
  const int ar = tid >> 1, ac = (tid & 1) * 16;     // A: row, col-of-16
  const int bk = tid >> 3, bn = (tid & 7) * 16;     // B: k-row, n-col-of-16
  const bf16* srcA = A + (size_t)(m_blk + ar) * K + ac;
  const bf16* srcB = B + (size_t)bk * N + n_blk + bn;
  const size_t bstep = (size_t)32 * N;

  v8f acc[8] = {};

  for (int k0 = 0; k0 < K; k0 += 32) {
    { // A tile: 32-byte vector copy straight into LDS
      const uint4* a4 = reinterpret_cast<const uint4*>(srcA);
      *reinterpret_cast<uint4*>(&As[ar][ac])     = a4[0];
      *reinterpret_cast<uint4*>(&As[ar][ac + 8]) = a4[1];
    }
    { // B tile: 32-byte vector load, transpose-scatter into Bt[n][k]
      const uint4* b4 = reinterpret_cast<const uint4*>(srcB);
      uint4 d0 = b4[0], d1 = b4[1];
      bf16 tmp[16];
      *reinterpret_cast<uint4*>(&tmp[0]) = d0;
      *reinterpret_cast<uint4*>(&tmp[8]) = d1;
#pragma unroll
      for (int i = 0; i < 16; i++) Bt[bn + i][bk] = tmp[i];
    }
    // prefetch next K-tiles into cache (global_prefetch_b8)
    __builtin_prefetch(srcA + 32, 0, 1);
    __builtin_prefetch(srcB + bstep, 0, 1);
    srcA += 32;
    srcB += bstep;
    __syncthreads();

    // --- preload ALL fragments, then run the WMMA burst ---
    // A fragment (16x32): lane -> M=m0+lrow; halves i -> K = hi*16 + i
    v16bf afrag;
    {
      const bf16* pa  = &As[m0 + lrow][hi * 16];
      const bf16* pa2 = &As[m0 + lrow][hi * 16 + 8];
#pragma unroll
      for (int i = 0; i < 8; i++) { afrag[i] = pa[i]; afrag[8 + i] = pa2[i]; }
    }
    // B fragments (32x16): lane -> N=nf*16+lrow; halves i -> K = hi*16 + i
    v16bf bfr[8];
#pragma unroll
    for (int nf = 0; nf < 8; nf++) {
      const bf16* pb = &Bt[nf * 16 + lrow][hi * 16];
#pragma unroll
      for (int i = 0; i < 16; i++) bfr[nf][i] = pb[i];
    }
#pragma unroll
    for (int nf = 0; nf < 8; nf++) {
      acc[nf] = __builtin_amdgcn_wmma_f32_16x16x32_bf16(
          false, afrag, false, bfr[nf], (short)0, acc[nf], false, false);
    }
    __syncthreads();
  }

  // C layout: lane -> N = n0+lrow; vgpr r -> M = m0 + r + hi*8
  if (mode == 0) {
    float* C = (float*)Cout;
#pragma unroll
    for (int nf = 0; nf < 8; nf++) {
      int n = n_blk + nf * 16 + lrow;
#pragma unroll
      for (int r = 0; r < 8; r++) {
        int m = m_blk + m0 + r + hi * 8;
        C[(size_t)m * N + n] = acc[nf][r];
      }
    }
  } else {
    bf16* C = (bf16*)Cout;
#pragma unroll
    for (int nf = 0; nf < 8; nf++) {
      int n = n_blk + nf * 16 + lrow;
#pragma unroll
      for (int r = 0; r < 8; r++) {
        int m = m_blk + m0 + r + hi * 8;
        float v = acc[nf][r];
        if (mode == 2) v = v / (1.0f + __expf(-v));   // SiLU
        C[(size_t)m * N + n] = (bf16)v;
      }
    }
  }
}

// =====================================================================
// compressed pooling: Kc/Vc[g][n][d] = sum_l phi[l]*k[(n*16+l)][g*64+d]
// =====================================================================
__global__ __launch_bounds__(64) void pool_kcvc_kernel(
    const bf16* __restrict__ Kb, const bf16* __restrict__ Vb,
    const float* __restrict__ phik, const float* __restrict__ phiv,
    float* __restrict__ Kc, float* __restrict__ Vc) {
  const int d = threadIdx.x;          // 0..63
  const int n = blockIdx.x;           // 0..126
  const int g = blockIdx.y;           // 0..3
  float ak = 0.f, av = 0.f;
  for (int l = 0; l < 32; l++) {
    size_t row = (size_t)(n * 16 + l);
    ak += phik[l] * (float)Kb[row * LDKV + g * DK + d];
    av += phiv[l] * (float)Vb[row * LDKV + g * DK + d];
  }
  Kc[((size_t)g * NC + n) * DK + d] = ak;
  Vc[((size_t)g * NC + n) * DK + d] = av;
}

// =====================================================================
// compressed attention: per (g,s): 4 heads, softmax over 127 blocks
// (causal full-block mask), o_cmp, plus pcsum = sum over heads.
// =====================================================================
__global__ __launch_bounds__(128) void cmp_attn_kernel(
    const bf16* __restrict__ Qb, const float* __restrict__ Kc,
    const float* __restrict__ Vc, float* __restrict__ Ocmp,
    float* __restrict__ pcsum) {
  __shared__ float sh[128];
  __shared__ float psh[128];
  const int tid = threadIdx.x;
  const int s = blockIdx.x;
  const int g = blockIdx.y;
  const int n = tid;
  const bool nvalid = (n < NC);
  const bool active = nvalid && (n * 16 + 31 <= s);
  float psum_acc = 0.f;

  for (int h = 0; h < HPG; h++) {
    const int head = g * HPG + h;
    float sc = -1e30f;
    if (active) {
      const bf16* qp = Qb + (size_t)s * LDQ + head * DK;
      const float* kp = Kc + ((size_t)g * NC + n) * DK;
      float acc = 0.f;
      for (int d = 0; d < DK; d++) acc += (float)qp[d] * kp[d];
      sc = acc * 0.125f;   // 1/sqrt(64)
    }
    sh[tid] = sc;
    __syncthreads();
    for (int off = 64; off > 0; off >>= 1) {
      if (tid < off) sh[tid] = fmaxf(sh[tid], sh[tid + off]);
      __syncthreads();
    }
    float mx = sh[0];
    __syncthreads();
    float p = (active && mx > -1e29f) ? __expf(sc - mx) : 0.f;
    sh[tid] = p;
    __syncthreads();
    for (int off = 64; off > 0; off >>= 1) {
      if (tid < off) sh[tid] += sh[tid + off];
      __syncthreads();
    }
    float denom = sh[0];
    __syncthreads();
    p = (denom > 0.f) ? p / denom : 0.f;
    psh[tid] = p;
    psum_acc += p;
    __syncthreads();
    if (tid < DK) {
      float o = 0.f;
      for (int nn = 0; nn < NC; nn++)
        o += psh[nn] * Vc[((size_t)g * NC + nn) * DK + tid];
      Ocmp[((size_t)head * S_LEN + s) * DK + tid] = o;
    }
    __syncthreads();
  }
  if (nvalid) pcsum[((size_t)g * S_LEN + s) * NC + n] = psum_acc;
}

// =====================================================================
// selection scoring + top-16 of 32 blocks -> bitmask per (g,s)
// =====================================================================
__global__ __launch_bounds__(32) void select_topk_kernel(
    const float* __restrict__ pcsum, unsigned int* __restrict__ selmask) {
  __shared__ float sc[32];
  const int m = threadIdx.x;
  const int s = blockIdx.x;
  const int g = blockIdx.y;
  const float* pc = pcsum + ((size_t)g * S_LEN + s) * NC;
  float acc = 0.f;
  for (int j = 0; j < NC; j++) {
    int s0 = j * 16, s1 = s0 + 32;
    int lo = max(s0, m * 64), hib = min(s1, m * 64 + 64);
    if (hib > lo) acc += pc[j] * (float)(hib - lo) * (1.0f / 32.0f);
  }
  const bool allowed = (m * 64 <= s);
  const bool forced  = (m == 0) || (m == (s >> 6));
  float score = allowed ? (acc + (forced ? 1e6f : 0.f)) : -1e9f;
  sc[m] = score;
  __syncthreads();
  int rank = 0;
  for (int j = 0; j < 32; j++) {
    float v = sc[j];
    if (v > score || (v == score && j < m)) rank++;
  }
  bool selected = allowed && (rank < NSEL);
  unsigned long long b = __ballot(selected);
  if (m == 0) selmask[(size_t)g * S_LEN + s] = (unsigned int)b;
}

// =====================================================================
// fused flash attention (WMMA) for select (branch=1) / window (branch=0)
// grid: (S/128, 16 heads), 256 threads = 8 waves, wave = 16 query rows.
// No __syncthreads: each wave uses its own LDS slice (wave-ordered LDS).
// =====================================================================
__global__ __launch_bounds__(256) void flash_attn_kernel(
    const bf16* __restrict__ Qb, const bf16* __restrict__ Kb,
    const bf16* __restrict__ Vb, const unsigned int* __restrict__ selmask,
    float* __restrict__ O, int branch) {
  __shared__ float s_s[8][16][32];
  __shared__ bf16  s_p[8][16][32];
  __shared__ float s_max[8][16];
  __shared__ float s_sum[8][16];
  __shared__ float s_fac[8][16];

  const int tid = threadIdx.x, wave = tid >> 5, lane = tid & 31;
  const int lrow = lane & 15, hi = lane >> 4;
  const int head = blockIdx.y;
  const int g = head >> 2;
  const int q0 = blockIdx.x * 128 + wave * 16;
  const float scale = 0.125f;

  // Q fragments for dk 0..31 and 32..63 (A layout)
  v16bf qf[2];
  {
    const bf16* qrow = Qb + (size_t)(q0 + lrow) * LDQ + head * DK;
#pragma unroll
    for (int f = 0; f < 2; f++) {
      const bf16* p1 = qrow + f * 32 + hi * 16;
      const bf16* p2 = qrow + f * 32 + hi * 16 + 8;
#pragma unroll
      for (int i = 0; i < 8; i++) { qf[f][i] = p1[i]; qf[f][8 + i] = p2[i]; }
    }
  }

  v8f oacc[4] = {};
  if (lane < 16) { s_max[wave][lane] = -1e30f; s_sum[wave][lane] = 0.f; }

  // per-row select masks for this lane's output rows, and wave union mask
  unsigned int rowmask[8];
  unsigned int umask = 0;
  if (branch == 1) {
#pragma unroll
    for (int r = 0; r < 8; r++)
      rowmask[r] = selmask[(size_t)g * S_LEN + q0 + r + hi * 8];
    unsigned int mrow = selmask[(size_t)g * S_LEN + q0 + lrow];
#pragma unroll
    for (int off = 1; off < 16; off <<= 1)
      mrow |= (unsigned int)__shfl_xor((int)mrow, off, 16);
    umask = mrow;   // uniform across wave
  } else {
#pragma unroll
    for (int r = 0; r < 8; r++) rowmask[r] = 0u;
  }

  auto process_chunk = [&](int j0) {
    // --- scores: S(16x32) = Q(16x64) @ K^T(64x32) ---
    v8f sfr[2];
#pragma unroll
    for (int c = 0; c < 2; c++) {
      v8f sacc = {};
#pragma unroll
      for (int f = 0; f < 2; f++) {
        v16bf kfrag;   // B frag 32x16: N=key lane, halves i -> kdim f*32+hi*16+i
        const bf16* kp =
            Kb + (size_t)(j0 + c * 16 + lrow) * LDKV + g * DK + f * 32 + hi * 16;
#pragma unroll
        for (int i = 0; i < 16; i++) kfrag[i] = kp[i];
        sacc = __builtin_amdgcn_wmma_f32_16x16x32_bf16(
            false, qf[f], false, kfrag, (short)0, sacc, false, false);
      }
      sfr[c] = sacc;
    }
    // --- mask in C-layout registers, spill to LDS ---
#pragma unroll
    for (int c = 0; c < 2; c++) {
#pragma unroll
      for (int r = 0; r < 8; r++) {
        int srow = q0 + r + hi * 8;
        int j = j0 + c * 16 + lrow;
        bool ok;
        if (branch == 0)
          ok = (j <= srow) && (j > srow - WWIN);
        else
          ok = (j <= srow) && ((rowmask[r] >> (j >> 6)) & 1u);
        s_s[wave][r + hi * 8][c * 16 + lrow] = ok ? sfr[c][r] * scale : -1e9f;
      }
    }
    // --- online softmax row pass (lanes 0..15, one row each) ---
    if (lane < 16) {
      float om = s_max[wave][lane];
      float cm = -1e30f;
#pragma unroll
      for (int j = 0; j < 32; j++) cm = fmaxf(cm, s_s[wave][lane][j]);
      float nm = fmaxf(om, cm);
      float fac = __expf(om - nm);
      float sum = 0.f;
#pragma unroll
      for (int j = 0; j < 32; j++) {
        float v = s_s[wave][lane][j];
        float p = (v > -1e8f) ? __expf(v - nm) : 0.f;
        s_p[wave][lane][j] = (bf16)p;
        sum += p;
      }
      s_max[wave][lane] = nm;
      s_sum[wave][lane] = s_sum[wave][lane] * fac + sum;
      s_fac[wave][lane] = fac;
    }
    // --- rescale O accumulators ---
#pragma unroll
    for (int f = 0; f < 4; f++)
#pragma unroll
      for (int r = 0; r < 8; r++) oacc[f][r] *= s_fac[wave][r + hi * 8];
    // --- O += P(16x32) @ V(32x64) ---
    v16bf pfrag;
    {
      const bf16* pp  = &s_p[wave][lrow][hi * 16];
      const bf16* pp2 = &s_p[wave][lrow][hi * 16 + 8];
#pragma unroll
      for (int i = 0; i < 8; i++) { pfrag[i] = pp[i]; pfrag[8 + i] = pp2[i]; }
    }
#pragma unroll
    for (int f = 0; f < 4; f++) {
      v16bf vfrag;   // B frag: N=d lane, halves i -> key j0+hi*16+i
#pragma unroll
      for (int i = 0; i < 16; i++)
        vfrag[i] = Vb[(size_t)(j0 + i + hi * 16) * LDKV + g * DK + f * 16 + lrow];
      oacc[f] = __builtin_amdgcn_wmma_f32_16x16x32_bf16(
          false, pfrag, false, vfrag, (short)0, oacc[f], false, false);
    }
  };

  if (branch == 0) {
    int jlo = q0 - (WWIN - 1);
    if (jlo < 0) jlo = 0;
    jlo &= ~31;
    int jhi = q0 + 15;
    for (int j0 = jlo; j0 <= jhi; j0 += 32) process_chunk(j0);
  } else {
    for (int m = 0; m < NB; m++) {
      if (!((umask >> m) & 1u)) continue;
      int base = m * 64;
      if (base > q0 + 15) continue;
      process_chunk(base);
      if (base + 32 <= q0 + 15) process_chunk(base + 32);
    }
  }

  // normalize + write O[head][s][d]
#pragma unroll
  for (int f = 0; f < 4; f++) {
#pragma unroll
    for (int r = 0; r < 8; r++) {
      int srow = q0 + r + hi * 8;
      float denom = s_sum[wave][r + hi * 8];
      float val = (denom > 0.f) ? oacc[f][r] / denom : 0.f;
      O[((size_t)head * S_LEN + srow) * DK + f * 16 + lrow] = val;
    }
  }
}

// =====================================================================
// gate merge: Om[s][head*64+d] = sum_c sigmoid(gate[s][head*3+c]) * O_c
// =====================================================================
__global__ void merge_gate_kernel(
    const float* __restrict__ Ocmp, const float* __restrict__ Osel,
    const float* __restrict__ Owin, const float* __restrict__ Gt,
    bf16* __restrict__ Om) {
  int idx = blockIdx.x * 256 + threadIdx.x;
  if (idx >= S_LEN * NHEAD * DK) return;
  int d = idx & 63;
  int head = (idx >> 6) & 15;
  int s = idx >> 10;
  size_t oidx = ((size_t)head * S_LEN + s) * DK + d;
  const float* gp = Gt + (size_t)s * GATE_LD + head * 3;
  float g0 = 1.f / (1.f + __expf(-gp[0]));
  float g1 = 1.f / (1.f + __expf(-gp[1]));
  float g2 = 1.f / (1.f + __expf(-gp[2]));
  float o = g0 * Ocmp[oidx] + g1 * Osel[oidx] + g2 * Owin[oidx];
  Om[(size_t)s * LDQ + head * DK + d] = (bf16)o;
}

// =====================================================================
// host launch
// =====================================================================
extern "C" void kernel_launch(void* const* d_in, const int* in_sizes, int n_in,
                              void* d_out, int out_size, void* d_ws, size_t ws_size,
                              hipStream_t stream) {
  const float* x      = (const float*)d_in[0];
  const float* Wq     = (const float*)d_in[1];
  const float* Wk     = (const float*)d_in[2];
  const float* Wv     = (const float*)d_in[3];
  const float* phi_k  = (const float*)d_in[4];
  const float* phi_v  = (const float*)d_in[5];
  const float* Wg     = (const float*)d_in[6];
  const float* Wo     = (const float*)d_in[7];
  const float* norm1w = (const float*)d_in[8];
  const float* norm2w = (const float*)d_in[9];
  const float* fc1    = (const float*)d_in[10];
  const float* fc2    = (const float*)d_in[11];
  float* out = (float*)d_out;

  char* wsp = (char*)d_ws;
  auto alloc = [&](size_t bytes) -> void* {
    void* p = (void*)wsp;
    wsp += (bytes + 255) & ~(size_t)255;
    return p;
  };

  bf16* xn_bf  = (bf16*)alloc((size_t)S_LEN * DIM * 2);
  bf16* wq_bf  = (bf16*)alloc((size_t)DIM * LDQ * 2);
  bf16* wk_bf  = (bf16*)alloc((size_t)DIM * LDKV * 2);
  bf16* wv_bf  = (bf16*)alloc((size_t)DIM * LDKV * 2);
  bf16* wg_bf  = (bf16*)alloc((size_t)DIM * GATE_LD * 2);   // padded
  bf16* wo_bf  = (bf16*)alloc((size_t)LDQ * DIM * 2);
  bf16* fc1_bf = (bf16*)alloc((size_t)DIM * 4 * DIM * 2);
  bf16* fc2_bf = (bf16*)alloc((size_t)4 * DIM * DIM * 2);
  bf16*  q_bf  = (bf16*) alloc((size_t)S_LEN * LDQ * 2);
  bf16*  k_bf  = (bf16*) alloc((size_t)S_LEN * LDKV * 2);
  bf16*  v_bf  = (bf16*) alloc((size_t)S_LEN * LDKV * 2);
  float* gates = (float*)alloc((size_t)S_LEN * GATE_LD * 4); // padded
  float* Kc    = (float*)alloc((size_t)NGRP * NC * DK * 4);
  float* Vc    = (float*)alloc((size_t)NGRP * NC * DK * 4);
  float* o_cmp = (float*)alloc((size_t)NHEAD * S_LEN * DK * 4);
  float* o_sel = (float*)alloc((size_t)NHEAD * S_LEN * DK * 4);
  float* o_win = (float*)alloc((size_t)NHEAD * S_LEN * DK * 4);
  float* pcsum = (float*)alloc((size_t)NGRP * S_LEN * NC * 4);
  unsigned int* selmask = (unsigned int*)alloc((size_t)NGRP * S_LEN * 4);
  bf16*  om_bf = (bf16*) alloc((size_t)S_LEN * LDQ * 2);
  float* attno = (float*)alloc((size_t)S_LEN * DIM * 4);
  float* x1    = (float*)alloc((size_t)S_LEN * DIM * 4);
  bf16*  hm_bf = (bf16*) alloc((size_t)S_LEN * DIM * 2);
  bf16*  h1_bf = (bf16*) alloc((size_t)S_LEN * 4 * DIM * 2);
  float* y_f   = (float*)alloc((size_t)S_LEN * DIM * 4);

  auto conv = [&](const float* s, bf16* d, int n) {
    f32_to_bf16_kernel<<<(n + 255) / 256, 256, 0, stream>>>(s, d, n);
  };
  auto gemm = [&](const bf16* A, const bf16* B, void* C, int M, int N, int K,
                  int mode) {
    dim3 grid(N / 128, M / 128);
    gemm_bf16_wmma<<<grid, 256, 0, stream>>>(A, B, C, M, N, K, mode);
  };

  // weight converts
  conv(Wq, wq_bf, DIM * LDQ);
  conv(Wk, wk_bf, DIM * LDKV);
  conv(Wv, wv_bf, DIM * LDKV);
  convpad_wg_kernel<<<(DIM * GATE_LD + 255) / 256, 256, 0, stream>>>(Wg, wg_bf);
  conv(Wo, wo_bf, LDQ * DIM);
  conv(fc1, fc1_bf, DIM * 4 * DIM);
  conv(fc2, fc2_bf, 4 * DIM * DIM);

  // norm1
  rmsnorm_kernel<<<S_LEN, 256, 0, stream>>>(x, nullptr, norm1w, nullptr, xn_bf, DIM);

  // projections (bf16 epilogue, no separate converts)
  gemm(xn_bf, wq_bf, q_bf, S_LEN, LDQ, DIM, 1);
  gemm(xn_bf, wk_bf, k_bf, S_LEN, LDKV, DIM, 1);
  gemm(xn_bf, wv_bf, v_bf, S_LEN, LDKV, DIM, 1);
  gemm(xn_bf, wg_bf, gates, S_LEN, GATE_LD, DIM, 0);

  // compressed branch
  pool_kcvc_kernel<<<dim3(NC, NGRP), 64, 0, stream>>>(k_bf, v_bf, phi_k, phi_v, Kc, Vc);
  cmp_attn_kernel<<<dim3(S_LEN, NGRP), 128, 0, stream>>>(q_bf, Kc, Vc, o_cmp, pcsum);

  // selection
  select_topk_kernel<<<dim3(S_LEN, NGRP), 32, 0, stream>>>(pcsum, selmask);

  // flash attention branches (WMMA)
  flash_attn_kernel<<<dim3(S_LEN / 128, NHEAD), 256, 0, stream>>>(
      q_bf, k_bf, v_bf, selmask, o_win, 0);
  flash_attn_kernel<<<dim3(S_LEN / 128, NHEAD), 256, 0, stream>>>(
      q_bf, k_bf, v_bf, selmask, o_sel, 1);

  // gate merge + output projection
  merge_gate_kernel<<<(S_LEN * NHEAD * DK + 255) / 256, 256, 0, stream>>>(
      o_cmp, o_sel, o_win, gates, om_bf);
  gemm(om_bf, wo_bf, attno, S_LEN, DIM, LDQ, 0);

  // residual + norm2 + MLP (SiLU fused into fc1 epilogue)
  rmsnorm_kernel<<<S_LEN, 256, 0, stream>>>(x, attno, norm2w, x1, hm_bf, DIM);
  gemm(hm_bf, fc1_bf, h1_bf, S_LEN, 4 * DIM, DIM, 2);
  gemm(h1_bf, fc2_bf, y_f, S_LEN, DIM, 4 * DIM, 0);
  add_out_kernel<<<(S_LEN * DIM + 255) / 256, 256, 0, stream>>>(
      x1, y_f, out, S_LEN * DIM);
}
